// GAT_53249004536374
// MI455X (gfx1250) — matrix operand order repaired and checked by reading, not compiled, not run on hardware
//
#include <hip/hip_runtime.h>
#include <cstddef>
#include <cstdint>

#define N_NODES 4096
#define F_IN    128
#define NHID    64
#define NHEADS  8
#define ALPHA   0.2f
#define MCOLS   128     // columns fetched per TDM tile (4 WMMA K-tiles)

typedef __attribute__((ext_vector_type(16))) _Float16     v16h;
typedef __attribute__((ext_vector_type(8)))  _Float16     v8h;
typedef __attribute__((ext_vector_type(8)))  float        v8f;
typedef __attribute__((ext_vector_type(4)))  unsigned int u32x4;
typedef __attribute__((ext_vector_type(8)))  int          i32x8;
typedef __attribute__((ext_vector_type(4)))  int          i32x4;

__device__ __forceinline__ float lrelu(float x) { return x > 0.f ? x : ALPHA * x; }

// ds_swizzle xor-shuffle within wave32 (imm must be a literal: (xor<<10)|0x1F)
#define SWZ(v, imm) __int_as_float(__builtin_amdgcn_ds_swizzle(__float_as_int(v), (imm)))

// Tensor Data Mover available? (host pass reports 0 -> falls back to plain loads)
#if __has_builtin(__builtin_amdgcn_tensor_load_to_lds) && __has_builtin(__builtin_amdgcn_s_wait_tensorcnt)
#define USE_TDM 1
#else
#define USE_TDM 0
#endif

#if USE_TDM
// Issue a TDM load of a 16x128 fp32 tile (pitch 4096 floats) of adj into LDS.
// D# packing per CDNA5 ISA section 8.3/8.4: group0 = {count|flags, lds_addr,
// global_addr[31:0], global_addr[56:32]|type=2}, group1 = dims/tile/stride.
__device__ __forceinline__ void tdm_load_adj_tile(const float* g, unsigned int lds_off)
{
    const unsigned long long ga = (unsigned long long)(uintptr_t)g;
    u32x4 g0 = {};
    g0.x = 1u;                                           // count=1 valid descriptor
    g0.y = lds_off;                                      // lds_addr (bytes)
    g0.z = (unsigned int)ga;                             // global_addr[31:0]
    g0.w = (unsigned int)((ga >> 32) & 0x01FFFFFFu)      // global_addr[56:32]
         | (2u << 30);                                   // type = 2 ("image")
    i32x8 g1 = {};
    g1[0] = 0x00020000;          // workgroup_mask=0, data_size=2 (4 bytes)
    g1[1] = 0x10000000;          // tensor_dim0 = 4096 (lo16 @ bits[63:48])
    g1[2] = 0x10000000;          // tensor_dim0 hi=0; tensor_dim1 = 4096 lo16
    g1[3] = (MCOLS << 16);       // tensor_dim1 hi=0; tile_dim0 = 128
    g1[4] = 16;                  // tile_dim1 = 16 rows; tile_dim2 = 0
    g1[5] = 4096;                // tensor_dim0_stride = 4096 elements
    g1[6] = 0;                   // stride hi / tensor_dim1_stride lo
    g1[7] = 0;
    i32x4 z4 = {};
#if defined(__clang_major__) && __clang_major__ >= 23
    i32x8 z8 = {};
    __builtin_amdgcn_tensor_load_to_lds(g0, g1, z4, z4, z8, 0);
#else
    __builtin_amdgcn_tensor_load_to_lds(g0, g1, z4, z4, 0);
#endif
}
#endif // USE_TDM

// ---------------------------------------------------------------------------
// Kernel 1: Wh = x @ W per head via v_wmma_f32_16x16x32_f16.
// Block = 128 threads (4 waves); wave w owns D-slice [16w,16w+16).
// Writes Wh fp32 [H][N][D] and WhT f16 [H][D][N] (B-operand-friendly layout).
// ---------------------------------------------------------------------------
__global__ void __launch_bounds__(128)
gat_wh_wmma(const float* __restrict__ x, const float* __restrict__ W,
            float* __restrict__ Wh, _Float16* __restrict__ WhT)
{
    const int lane = threadIdx.x & 31;
    const int wave = threadIdx.x >> 5;      // d-tile index 0..3
    const int blk  = blockIdx.x;            // h*256 + rowblock
    const int h    = blk >> 8;
    const int rb   = blk & 255;

    const int half = lane >> 4;             // A: 0 -> K 0-15, 1 -> K 16-31
    const int m    = lane & 15;             // A row / B,C column within tile
    const int kh   = half << 3;

    const float* xr = x + (size_t)(rb * 16 + m) * F_IN;

    v8f acc = {};
    #pragma unroll
    for (int kt = 0; kt < 4; ++kt) {
        const int k0 = kt * 32;
        alignas(16) float buf[16];
        *(float4*)&buf[0]  = *(const float4*)(xr + k0 + 16 * half);
        *(float4*)&buf[4]  = *(const float4*)(xr + k0 + 16 * half + 4);
        *(float4*)&buf[8]  = *(const float4*)(xr + k0 + 16 * half + 8);
        *(float4*)&buf[12] = *(const float4*)(xr + k0 + 16 * half + 12);
        v16h a;
        #pragma unroll
        for (int i = 0; i < 16; ++i) a[i] = (_Float16)buf[i];

        // B layout: lane%16 = N col (d), components = 16 consecutive K rows
        const float* wp = W + ((size_t)h * F_IN + k0 + half * 16) * NHID + wave * 16 + m;
        v16h b;
        #pragma unroll
        for (int i = 0; i < 16; ++i) b[i] = (_Float16)wp[(size_t)i * NHID];

        acc = __builtin_amdgcn_wmma_f32_16x16x32_f16(false, a, false, b,
                                                     (short)0, acc, false, false);
    }

    // C layout: VGPR j -> row (j + 8*half), lane%16 -> d column
    const int d = wave * 16 + m;
    v8h hh;
    #pragma unroll
    for (int j = 0; j < 8; ++j) {
        const int row = rb * 16 + j + 8 * half;
        Wh[((size_t)h * N_NODES + row) * NHID + d] = acc[j];
        hh[j] = (_Float16)acc[j];
    }
    *(v8h*)(WhT + ((size_t)h * NHID + d) * N_NODES + rb * 16 + 8 * half) = hh;
}

// ---------------------------------------------------------------------------
// Kernel 2: e_src[h][n] = Wh[h][n]·a1[h], e_dst likewise. One thread per (h,n).
// ---------------------------------------------------------------------------
__global__ void __launch_bounds__(256)
gat_edge_logits(const float* __restrict__ Wh, const float* __restrict__ a1,
                const float* __restrict__ a2, float* __restrict__ esrc,
                float* __restrict__ edst)
{
    const int idx = blockIdx.x * 256 + threadIdx.x;      // h*N + n
    if (idx >= NHEADS * N_NODES) return;
    const int h = idx >> 12;
    const int n = idx & (N_NODES - 1);
    const float* wr = Wh + ((size_t)h * N_NODES + n) * NHID;
    const float* p1 = a1 + h * NHID;
    const float* p2 = a2 + h * NHID;
    float s1 = 0.f, s2 = 0.f;
    #pragma unroll 8
    for (int d = 0; d < NHID; ++d) { const float v = wr[d]; s1 += v * p1[d]; s2 += v * p2[d]; }
    esrc[idx] = s1;
    edst[idx] = s2;
}

// ---------------------------------------------------------------------------
// Kernel 3: masked-softmax row statistics. One wave32 per (h, n).
// Online max/sum per lane over the masked row, then xor-tree wave reduction.
// ---------------------------------------------------------------------------
__global__ void __launch_bounds__(128)
gat_softmax_stats(const float* __restrict__ adj, const float* __restrict__ esrc,
                  const float* __restrict__ edst, float* __restrict__ rmax,
                  float* __restrict__ rinv)
{
    const int lane = threadIdx.x & 31;
    const int wid  = blockIdx.x * 4 + (threadIdx.x >> 5);   // h*N + n
    const int h = wid >> 12;
    const int n = wid & (N_NODES - 1);
    const float* adjr = adj + (size_t)n * N_NODES;
    const float* edr  = edst + (size_t)h * N_NODES;
    const float  es   = esrc[wid];

    float mx = -INFINITY, sum = 0.f;
    for (int it = 0; it < N_NODES / 128; ++it) {
        const int c = it * 128 + lane * 4;
        if (it + 4 < N_NODES / 128)                  // -> global_prefetch_b8
            __builtin_prefetch(adjr + c + 512, 0, 0);
        const float4 av = *(const float4*)(adjr + c);
        const float4 ev = *(const float4*)(edr + c);
        const float aa[4] = {av.x, av.y, av.z, av.w};
        const float ee[4] = {ev.x, ev.y, ev.z, ev.w};
        #pragma unroll
        for (int j = 0; j < 4; ++j) {
            if (aa[j] > 0.f) {
                const float e = lrelu(es + ee[j]);
                if (e > mx) { sum *= __expf(mx - e); mx = e; }
                sum += __expf(e - mx);
            }
        }
    }
    // wave32 xor-tree reduction of (max, sum); guard -inf lanes (sparse rows)
    #define COMBINE(IMM)                                                       \
    {                                                                          \
        const float om = SWZ(mx, IMM), os = SWZ(sum, IMM);                     \
        const float M = fmaxf(mx, om);                                         \
        float s = 0.f;                                                         \
        if (mx > -INFINITY) s += sum * __expf(mx - M);                         \
        if (om > -INFINITY) s += os  * __expf(om - M);                         \
        sum = s; mx = M;                                                       \
    }
    COMBINE(0x041F) COMBINE(0x081F) COMBINE(0x101F) COMBINE(0x201F) COMBINE(0x401F)
    #undef COMBINE

    if (lane == 0) { rmax[wid] = mx; rinv[wid] = 1.f / sum; }   // self-loop => sum > 0
}

// ---------------------------------------------------------------------------
// Kernel 4: fused attention aggregation h = ELU(softmax(e) @ Wh), head-concat.
// One wave per (h, 16-row block). adj tiles (16x128 fp32, wave-uniform 2D
// tiles) are double-buffered into LDS by the Tensor Data Mover; P tile built
// in registers in WMMA A-layout; 4x v_wmma_f32_16x16x32_f16 per 32-col K-tile.
// ---------------------------------------------------------------------------
__global__ void __launch_bounds__(128)
gat_attn_aggregate(const float* __restrict__ adj, const float* __restrict__ esrc,
                   const float* __restrict__ rmax, const float* __restrict__ rinv,
                   const float* __restrict__ edst, const _Float16* __restrict__ WhT,
                   float* __restrict__ out)
{
    __shared__ float sED[N_NODES];                 // 16 KB
#if USE_TDM
    __shared__ float sAdj[4 * 2 * 16 * MCOLS];     // 4 waves x 2 bufs x 8 KB = 64 KB
#endif
    const int lane = threadIdx.x & 31;
    const int wave = threadIdx.x >> 5;
    const int wid  = blockIdx.x * 4 + wave;        // h*256 + rowblock
    const int h    = wid >> 8;
    const int rb   = wid & 255;

    {   // cooperative stage of e_dst[h][0..4095] (all 4 waves share head h)
        const float* src = edst + (size_t)h * N_NODES;
        for (int i = threadIdx.x * 4; i < N_NODES; i += 128 * 4)
            *(float4*)&sED[i] = *(const float4*)(src + i);
    }
    __syncthreads();

    const int half = lane >> 4;
    const int m    = lane & 15;
    const int kh   = half << 4;
    const int rowg = rb * 16 + m;

    const float es = esrc[(size_t)h * N_NODES + rowg];
    const float mx = rmax[(size_t)h * N_NODES + rowg];
    const float ri = rinv[(size_t)h * N_NODES + rowg];
    const float* adjr = adj + (size_t)rowg * N_NODES;       // fallback path
    const float* adjw = adj + (size_t)(rb * 16) * N_NODES;  // wave tile origin

    v8f acc0 = {}, acc1 = {}, acc2 = {}, acc3 = {};

#if USE_TDM
    float* buf0 = &sAdj[(wave * 2 + 0) * 16 * MCOLS];
    float* buf1 = &sAdj[(wave * 2 + 1) * 16 * MCOLS];
    const unsigned int off0 = (unsigned int)(uintptr_t)buf0;  // generic addr low 32b = LDS byte offset
    const unsigned int off1 = (unsigned int)(uintptr_t)buf1;
    tdm_load_adj_tile(adjw, off0);                            // prologue: tile 0
#endif

    for (int mt = 0; mt < N_NODES / MCOLS; ++mt) {
#if USE_TDM
        if (mt + 1 < N_NODES / MCOLS) {
            tdm_load_adj_tile(adjw + (size_t)(mt + 1) * MCOLS, (mt & 1) ? off0 : off1);
            __builtin_amdgcn_s_wait_tensorcnt(1);   // current buffer complete
        } else {
            __builtin_amdgcn_s_wait_tensorcnt(0);
        }
        const float* cur = (mt & 1) ? buf1 : buf0;  // row-major 16 x 128 tile
#endif
        #pragma unroll
        for (int t4 = 0; t4 < MCOLS / 32; ++t4) {
            const int lc = t4 * 32 + kh;            // column within macro tile
            const int c0 = mt * MCOLS + lc;         // global column
            alignas(16) float ab[16], eb[16];
#if USE_TDM
            const float* ar = cur + m * MCOLS;      // my row in the LDS tile
            *(float4*)&ab[0]  = *(const float4*)(ar + lc);
            *(float4*)&ab[4]  = *(const float4*)(ar + lc + 4);
            *(float4*)&ab[8]  = *(const float4*)(ar + lc + 8);
            *(float4*)&ab[12] = *(const float4*)(ar + lc + 12);
#else
            *(float4*)&ab[0]  = *(const float4*)(adjr + c0);
            *(float4*)&ab[4]  = *(const float4*)(adjr + c0 + 4);
            *(float4*)&ab[8]  = *(const float4*)(adjr + c0 + 8);
            *(float4*)&ab[12] = *(const float4*)(adjr + c0 + 12);
#endif
            *(float4*)&eb[0]  = *(const float4*)&sED[c0];
            *(float4*)&eb[4]  = *(const float4*)&sED[c0 + 4];
            *(float4*)&eb[8]  = *(const float4*)&sED[c0 + 8];
            *(float4*)&eb[12] = *(const float4*)&sED[c0 + 12];

            v16h a;
            #pragma unroll
            for (int i = 0; i < 16; ++i) {
                const float e = lrelu(es + eb[i]);
                const float p = __expf(e - mx) * ri;     // normalized attention
                a[i] = (_Float16)(ab[i] > 0.f ? p : 0.f);
            }

            // B operand: WhT[h][dt*16+m][c0base .. +15] -> contiguous 32B loads
            const _Float16* bb = WhT + ((size_t)h * NHID + m) * N_NODES
                                     + (size_t)(mt * MCOLS + t4 * 32) + half * 16;
            const v16h b0 = *(const v16h*)(bb);
            const v16h b1 = *(const v16h*)(bb + (size_t)16 * N_NODES);
            const v16h b2 = *(const v16h*)(bb + (size_t)32 * N_NODES);
            const v16h b3 = *(const v16h*)(bb + (size_t)48 * N_NODES);

            acc0 = __builtin_amdgcn_wmma_f32_16x16x32_f16(false, a, false, b0, (short)0, acc0, false, false);
            acc1 = __builtin_amdgcn_wmma_f32_16x16x32_f16(false, a, false, b1, (short)0, acc1, false, false);
            acc2 = __builtin_amdgcn_wmma_f32_16x16x32_f16(false, a, false, b2, (short)0, acc2, false, false);
            acc3 = __builtin_amdgcn_wmma_f32_16x16x32_f16(false, a, false, b3, (short)0, acc3, false, false);
        }
    }

    // ELU + head-concat store: out[row][h*64 + d]
    #pragma unroll
    for (int j = 0; j < 8; ++j) {
        const int row = rb * 16 + j + 8 * half;
        float* orow = out + (size_t)row * (NHEADS * NHID) + h * NHID + m;
        const float v0 = acc0[j], v1 = acc1[j], v2 = acc2[j], v3 = acc3[j];
        orow[0]  = v0 > 0.f ? v0 : expm1f(v0);
        orow[16] = v1 > 0.f ? v1 : expm1f(v1);
        orow[32] = v2 > 0.f ? v2 : expm1f(v2);
        orow[48] = v3 > 0.f ? v3 : expm1f(v3);
    }
}

// ---------------------------------------------------------------------------
extern "C" void kernel_launch(void* const* d_in, const int* in_sizes, int n_in,
                              void* d_out, int out_size, void* d_ws, size_t ws_size,
                              hipStream_t stream)
{
    const float* x   = (const float*)d_in[0];
    const float* adj = (const float*)d_in[1];
    const float* W   = (const float*)d_in[2];
    const float* a1  = (const float*)d_in[3];
    const float* a2  = (const float*)d_in[4];
    float* out = (float*)d_out;

    char* ws = (char*)d_ws;
    float*    Wh   = (float*)ws;    ws += sizeof(float)    * NHEADS * N_NODES * NHID;
    _Float16* WhT  = (_Float16*)ws; ws += sizeof(_Float16) * NHEADS * NHID * N_NODES;
    float*    esrc = (float*)ws;    ws += sizeof(float)    * NHEADS * N_NODES;
    float*    edst = (float*)ws;    ws += sizeof(float)    * NHEADS * N_NODES;
    float*    rmax = (float*)ws;    ws += sizeof(float)    * NHEADS * N_NODES;
    float*    rinv = (float*)ws;    ws += sizeof(float)    * NHEADS * N_NODES;
    (void)in_sizes; (void)n_in; (void)out_size; (void)ws_size;

    gat_wh_wmma       <<<NHEADS * (N_NODES / 16),      128, 0, stream>>>(x, W, Wh, WhT);
    gat_edge_logits   <<<(NHEADS * N_NODES) / 256,     256, 0, stream>>>(Wh, a1, a2, esrc, edst);
    gat_softmax_stats <<<(NHEADS * N_NODES) / 4,       128, 0, stream>>>(adj, esrc, edst, rmax, rinv);
    gat_attn_aggregate<<<(NHEADS * (N_NODES / 16)) / 4, 128, 0, stream>>>(adj, esrc, rmax, rinv, edst, WhT, out);
}
